// FeedForward_21784074125378
// MI455X (gfx1250) — compile-verified, matrix operand order checked by reading, not run on hardware
//
#include <hip/hip_runtime.h>

// ---------------------------------------------------------------------------
// SwiGLU FFN with int8-range quantized weights, MI455X (gfx1250, wave32).
//   gate = x @ wg^T * sg ; up = x @ wu^T * su ; mid = silu(gate)*up
//   out  = mid @ wd^T * sd
// Pipeline: pre-convert weights (lossless int8->f16) and x (f32->f16) into
// workspace, then two WMMA f16 GEMM kernels using CDNA5 async-to-LDS staging
// (global_load_async_to_lds_b128 + ASYNCcnt) with double buffering.
// ---------------------------------------------------------------------------

typedef __attribute__((ext_vector_type(16))) _Float16 v16h;
typedef __attribute__((ext_vector_type(8)))  _Float16 v8h;
typedef __attribute__((ext_vector_type(4)))  _Float16 v4h;
typedef __attribute__((ext_vector_type(8)))  float    v8f;

#define FF_DIM    4096
#define FF_HID    11008
#define FF_M      8192          // B*S = 4*2048

// GEMM tiling: block 256 threads = 8 waves (4 M-waves x 2 N-waves),
// block tile 128x64, wave tile 32x32, K-stage 128 (4 WMMA K-steps).
#define BM   128
#define BN   64
#define BK   128
#define PAD  8                  // halfs; row stride 272B -> 4-bank rotation/row
#define LDT  (BK + PAD)         // 136 halfs per LDS tile row

#define A_BUF  (BM * LDT)       // 17408 halfs per A buffer
#define B_BUF  (BN * LDT)       // 8704 halfs per B buffer

// gate/up LDS layout (halfs): A[2] | Bg[2] | Bu[2]
#define GU_BG_OFF  (2 * A_BUF)             // 34816
#define GU_BU_OFF  (GU_BG_OFF + 2 * B_BUF) // 52224
#define GU_SMEM_BYTES ((GU_BU_OFF + 2 * B_BUF) * 2)  // 139264 B

// down LDS layout (halfs): A[2] | B[2]
#define DN_B_OFF   (2 * A_BUF)             // 34816
#define DN_SMEM_BYTES ((DN_B_OFF + 2 * B_BUF) * 2)   // 104448 B

// ---- CDNA5 async copy (global -> LDS), tracked by ASYNCcnt ----------------
__device__ __forceinline__ void async_ld_b128(unsigned lds_byte, const void* gaddr) {
  asm volatile("global_load_async_to_lds_b128 %0, %1, off"
               :: "v"(lds_byte), "v"(gaddr) : "memory");
}
__device__ __forceinline__ void wait_async0() {
  asm volatile("s_wait_asynccnt 0" ::: "memory");
}

// ---- fragment helpers -----------------------------------------------------
// 16-bit A/B fragment, 16x32 (rows x K) per ISA 7.12.2: lane holds row
// (lane&15), two contiguous 8-half (16B) chunks at K = (lane>>4)*8 and +16.
__device__ __forceinline__ v16h ldfrag(const _Float16* p) {
  v8h lo = *(const v8h*)(p);
  v8h hi = *(const v8h*)(p + 16);
  return __builtin_shufflevector(lo, hi, 0,1,2,3,4,5,6,7,8,9,10,11,12,13,14,15);
}

__device__ __forceinline__ v8f wmma16(v16h a, v16h b, v8f c) {
  return __builtin_amdgcn_wmma_f32_16x16x32_f16(
      false, a, false, b, (short)0, c, false, false);
}

// ---- conversion kernels ---------------------------------------------------
__global__ __launch_bounds__(256)
void cvt_i32_to_f16(const int* __restrict__ in, _Float16* __restrict__ out, int n4) {
  int i = blockIdx.x * blockDim.x + threadIdx.x;
  if (i < n4) {
    const int4 v = ((const int4*)in)[i];
    v4h o;
    o[0] = (_Float16)v.x; o[1] = (_Float16)v.y;
    o[2] = (_Float16)v.z; o[3] = (_Float16)v.w;
    ((v4h*)out)[i] = o;
  }
}

__global__ __launch_bounds__(256)
void cvt_f32_to_f16(const float* __restrict__ in, _Float16* __restrict__ out, int n4) {
  int i = blockIdx.x * blockDim.x + threadIdx.x;
  if (i < n4) {
    const float4 v = ((const float4*)in)[i];
    v4h o;
    o[0] = (_Float16)v.x; o[1] = (_Float16)v.y;
    o[2] = (_Float16)v.z; o[3] = (_Float16)v.w;
    ((v4h*)out)[i] = o;
  }
}

// ---- stage helpers: cooperative 256-thread async tile copies --------------
// A tile: BM(128) rows x BK(128) halfs = 2048 16B-chunks -> 8 per thread.
__device__ __forceinline__ void stage_tile_A(unsigned ldsBase, int buf,
                                             const _Float16* __restrict__ A,
                                             size_t lda, int rowBase, int kBase, int t) {
#pragma unroll
  for (int i = 0; i < 8; ++i) {
    int c = i * 256 + t;
    int r = c >> 4, kc = c & 15;
    const void* g = A + (size_t)(rowBase + r) * lda + kBase + kc * 8;
    unsigned l = ldsBase + (unsigned)(buf * A_BUF + r * LDT + kc * 8) * 2u;
    async_ld_b128(l, g);
  }
}
// B tile: BN(64) rows x BK(128) halfs = 1024 chunks -> 4 per thread.
__device__ __forceinline__ void stage_tile_B(unsigned ldsBase, unsigned off, int buf,
                                             const _Float16* __restrict__ B,
                                             size_t ldb, int rowBase, int kBase, int t) {
#pragma unroll
  for (int i = 0; i < 4; ++i) {
    int c = i * 256 + t;
    int r = c >> 4, kc = c & 15;
    const void* g = B + (size_t)(rowBase + r) * ldb + kBase + kc * 8;
    unsigned l = ldsBase + (off + (unsigned)(buf * B_BUF + r * LDT + kc * 8)) * 2u;
    async_ld_b128(l, g);
  }
}

// ---- fused gate/up GEMM + SwiGLU epilogue ---------------------------------
// grid: (M/128, HID/64), block 256.
__global__ __launch_bounds__(256)
void gate_up_kernel(const _Float16* __restrict__ x,   // [M, DIM]
                    const _Float16* __restrict__ wg,  // [HID, DIM]
                    const _Float16* __restrict__ wu,  // [HID, DIM]
                    const float* __restrict__ sgp, const float* __restrict__ sup,
                    _Float16* __restrict__ mid) {     // [M, HID]
  extern __shared__ __align__(16) _Float16 smem[];
  const unsigned ldsBase = (unsigned)(size_t)smem;   // LDS byte address (low 32b)

  const int t    = threadIdx.x;
  const int wid  = t >> 5;
  const int lane = t & 31;
  const int ln16 = lane & 15;
  const int g    = lane >> 4;
  const int mW   = (wid & 3) * 32;                   // wave M origin in tile
  const int nW   = (wid >> 2) * 32;                  // wave N origin in tile
  const int mBlock = blockIdx.x * BM;
  const int nBlock = blockIdx.y * BN;

  const float sg = sgp[0];
  const float su = sup[0];

  v8f accg[2][2] = {};
  v8f accu[2][2] = {};

  // prologue: stage buffer 0
  stage_tile_A(ldsBase, 0, x, FF_DIM, mBlock, 0, t);
  stage_tile_B(ldsBase, GU_BG_OFF, 0, wg, FF_DIM, nBlock, 0, t);
  stage_tile_B(ldsBase, GU_BU_OFF, 0, wu, FF_DIM, nBlock, 0, t);
  wait_async0();
  __syncthreads();

  const int nStages = FF_DIM / BK;   // 32
  for (int s = 0; s < nStages; ++s) {
    const int buf = s & 1;
    if (s + 1 < nStages) {
      const int kN = (s + 1) * BK;
      stage_tile_A(ldsBase, buf ^ 1, x, FF_DIM, mBlock, kN, t);
      stage_tile_B(ldsBase, GU_BG_OFF, buf ^ 1, wg, FF_DIM, nBlock, kN, t);
      stage_tile_B(ldsBase, GU_BU_OFF, buf ^ 1, wu, FF_DIM, nBlock, kN, t);
    }
    const _Float16* sA = smem + buf * A_BUF;
    const _Float16* sG = smem + GU_BG_OFF + buf * B_BUF;
    const _Float16* sU = smem + GU_BU_OFF + buf * B_BUF;
#pragma unroll
    for (int ks = 0; ks < BK / 32; ++ks) {
      const int ko = ks * 32 + g * 8;
      v16h a[2], bg[2], bu[2];
#pragma unroll
      for (int mi = 0; mi < 2; ++mi)
        a[mi] = ldfrag(sA + (mW + 16 * mi + ln16) * LDT + ko);
#pragma unroll
      for (int ni = 0; ni < 2; ++ni) {
        bg[ni] = ldfrag(sG + (nW + 16 * ni + ln16) * LDT + ko);
        bu[ni] = ldfrag(sU + (nW + 16 * ni + ln16) * LDT + ko);
      }
#pragma unroll
      for (int mi = 0; mi < 2; ++mi)
#pragma unroll
        for (int ni = 0; ni < 2; ++ni) {
          accg[mi][ni] = wmma16(a[mi], bg[ni], accg[mi][ni]);
          accu[mi][ni] = wmma16(a[mi], bu[ni], accu[mi][ni]);
        }
    }
    wait_async0();       // next-stage copies (this wave) complete
    __syncthreads();     // all waves done reading buf & all copies visible
  }

  // epilogue: mid = silu(g*sg) * (u*su).  C/D layout: VGPR j -> M=8*g+j, N=ln16
#pragma unroll
  for (int mi = 0; mi < 2; ++mi) {
#pragma unroll
    for (int ni = 0; ni < 2; ++ni) {
      const int r0  = mBlock + mW + 16 * mi + g * 8;
      const int col = nBlock + nW + 16 * ni + ln16;
#pragma unroll
      for (int j = 0; j < 8; ++j) {
        float gv = accg[mi][ni][j] * sg;
        float uv = accu[mi][ni][j] * su;
        float mv = gv * __builtin_amdgcn_rcpf(1.0f + __expf(-gv)) * uv;
        mid[(size_t)(r0 + j) * FF_HID + col] = (_Float16)mv;
      }
    }
  }
}

// ---- down-projection GEMM -------------------------------------------------
// grid: (M/128, DIM/64), block 256.
__global__ __launch_bounds__(256)
void down_kernel(const _Float16* __restrict__ mid,  // [M, HID]
                 const _Float16* __restrict__ wd,   // [DIM, HID]
                 const float* __restrict__ sdp,
                 float* __restrict__ out) {          // [M, DIM]
  extern __shared__ __align__(16) _Float16 smem[];
  const unsigned ldsBase = (unsigned)(size_t)smem;

  const int t    = threadIdx.x;
  const int wid  = t >> 5;
  const int lane = t & 31;
  const int ln16 = lane & 15;
  const int g    = lane >> 4;
  const int mW   = (wid & 3) * 32;
  const int nW   = (wid >> 2) * 32;
  const int mBlock = blockIdx.x * BM;
  const int nBlock = blockIdx.y * BN;

  const float sd = sdp[0];

  v8f acc[2][2] = {};

  stage_tile_A(ldsBase, 0, mid, FF_HID, mBlock, 0, t);
  stage_tile_B(ldsBase, DN_B_OFF, 0, wd, FF_HID, nBlock, 0, t);
  wait_async0();
  __syncthreads();

  const int nStages = FF_HID / BK;   // 86
  for (int s = 0; s < nStages; ++s) {
    const int buf = s & 1;
    if (s + 1 < nStages) {
      const int kN = (s + 1) * BK;
      stage_tile_A(ldsBase, buf ^ 1, mid, FF_HID, mBlock, kN, t);
      stage_tile_B(ldsBase, DN_B_OFF, buf ^ 1, wd, FF_HID, nBlock, kN, t);
    }
    const _Float16* sA = smem + buf * A_BUF;
    const _Float16* sB = smem + DN_B_OFF + buf * B_BUF;
#pragma unroll
    for (int ks = 0; ks < BK / 32; ++ks) {
      const int ko = ks * 32 + g * 8;
      v16h a[2], b[2];
#pragma unroll
      for (int mi = 0; mi < 2; ++mi)
        a[mi] = ldfrag(sA + (mW + 16 * mi + ln16) * LDT + ko);
#pragma unroll
      for (int ni = 0; ni < 2; ++ni)
        b[ni] = ldfrag(sB + (nW + 16 * ni + ln16) * LDT + ko);
#pragma unroll
      for (int mi = 0; mi < 2; ++mi)
#pragma unroll
        for (int ni = 0; ni < 2; ++ni)
          acc[mi][ni] = wmma16(a[mi], b[ni], acc[mi][ni]);
    }
    wait_async0();
    __syncthreads();
  }

#pragma unroll
  for (int mi = 0; mi < 2; ++mi) {
#pragma unroll
    for (int ni = 0; ni < 2; ++ni) {
      const int r0  = mBlock + mW + 16 * mi + g * 8;
      const int col = nBlock + nW + 16 * ni + ln16;
#pragma unroll
      for (int j = 0; j < 8; ++j)
        out[(size_t)(r0 + j) * FF_DIM + col] = acc[mi][ni][j] * sd;
    }
  }
}

// ---------------------------------------------------------------------------
extern "C" void kernel_launch(void* const* d_in, const int* in_sizes, int n_in,
                              void* d_out, int out_size, void* d_ws, size_t ws_size,
                              hipStream_t stream) {
  const float* x       = (const float*)d_in[0];
  const int*   w_gate  = (const int*)d_in[1];
  const int*   w_up    = (const int*)d_in[2];
  const int*   w_down  = (const int*)d_in[3];
  const float* s_gate  = (const float*)d_in[4];
  const float* s_up    = (const float*)d_in[5];
  const float* s_down  = (const float*)d_in[6];
  float*       out     = (float*)d_out;

  const size_t W_ELEMS   = (size_t)FF_HID * FF_DIM;   // 45,088,768
  const size_t X_ELEMS   = (size_t)FF_M   * FF_DIM;   // 33,554,432
  const size_t MID_ELEMS = (size_t)FF_M   * FF_HID;   // 90,177,536

  char* ws = (char*)d_ws;
  size_t off = 0;
  auto take = [&](size_t bytes) {
    char* p = ws + off;
    off += (bytes + 255) & ~(size_t)255;
    return p;
  };
  _Float16* wg_h  = (_Float16*)take(W_ELEMS * 2);
  _Float16* wu_h  = (_Float16*)take(W_ELEMS * 2);
  _Float16* wd_h  = (_Float16*)take(W_ELEMS * 2);
  _Float16* x_h   = (_Float16*)take(X_ELEMS * 2);
  _Float16* mid_h = (_Float16*)take(MID_ELEMS * 2);

  // allow >64KB dynamic LDS (host-side attribute set; capture-safe, idempotent)
  (void)hipFuncSetAttribute((const void*)gate_up_kernel,
                            hipFuncAttributeMaxDynamicSharedMemorySize, GU_SMEM_BYTES);
  (void)hipFuncSetAttribute((const void*)down_kernel,
                            hipFuncAttributeMaxDynamicSharedMemorySize, DN_SMEM_BYTES);

  // --- conversions (lossless for int8-valued weights) ---
  {
    const int n4w = (int)(W_ELEMS / 4);
    const int n4x = (int)(X_ELEMS / 4);
    dim3 blk(256);
    cvt_i32_to_f16<<<dim3((n4w + 255) / 256), blk, 0, stream>>>(w_gate, wg_h, n4w);
    cvt_i32_to_f16<<<dim3((n4w + 255) / 256), blk, 0, stream>>>(w_up,   wu_h, n4w);
    cvt_i32_to_f16<<<dim3((n4w + 255) / 256), blk, 0, stream>>>(w_down, wd_h, n4w);
    cvt_f32_to_f16<<<dim3((n4x + 255) / 256), blk, 0, stream>>>(x,      x_h,  n4x);
  }

  // --- fused gate/up + SwiGLU ---
  {
    dim3 grid(FF_M / BM, FF_HID / BN);
    gate_up_kernel<<<grid, dim3(256), GU_SMEM_BYTES, stream>>>(
        x_h, wg_h, wu_h, s_gate, s_up, mid_h);
  }

  // --- down projection ---
  {
    dim3 grid(FF_M / BM, FF_DIM / BN);
    down_kernel<<<grid, dim3(256), DN_SMEM_BYTES, stream>>>(mid_h, wd_h, s_down, out);
  }

  (void)in_sizes; (void)n_in; (void)out_size; (void)ws_size;
}
